// BiLevelRoutingAttention_66907000537867
// MI455X (gfx1250) — matrix operand order repaired and checked by reading, not compiled, not run on hardware
//
#include <hip/hip_runtime.h>

// ---------------------------------------------------------------------------
// BiLevel Routing Attention for MI455X (gfx1250, wave32, WMMA).
// All matmul stages use v_wmma_f32_16x16x32_bf16 (K=32 == head dim).
// Workload ~20 GFLOP, working set < L2 (192MB) -> matrix-pipe / load-issue
// bound, so: bf16 WMMA with fp32 accumulation, and 32x32 output tiles per
// wave in the dense GEMMs (2 A-frags x 2 B-frags -> 4 WMMAs per K-step) to
// halve fragment bytes loaded per WMMA.
// ---------------------------------------------------------------------------

#define DIMC  256
#define IMG   112
#define P2C   49
#define NPIX  (IMG*IMG)      // 12544 = 49*256

typedef __attribute__((ext_vector_type(16))) __bf16 v16bf;
typedef __attribute__((ext_vector_type(8)))  float  v8f;

union Frag32B { uint4 u[2]; v16bf v; };

__device__ __forceinline__ unsigned short f2bf(float f) {
    unsigned int u = __float_as_uint(f);
    unsigned int r = (u + 0x7FFFu + ((u >> 16) & 1u)) >> 16;
    return (unsigned short)r;
}
__device__ __forceinline__ float bf2f(unsigned short s) {
    return __uint_as_float(((unsigned int)s) << 16);
}

__device__ __forceinline__ v8f wmma_bf16(v16bf a, v16bf b, v8f c) {
    return __builtin_amdgcn_wmma_f32_16x16x32_bf16(false, a, false, b,
                                                   (short)0, c, false, false);
}

// A-fragment (16x32, M x K): lane l -> row m=l&15; K chunks {kb..kb+7, kb+16..kb+23},
// kb = (l>=16)?8:0.  src row-major with leading dim `ld` (elements).
__device__ __forceinline__ v16bf load_fragA(const unsigned short* base, int ld, int k0) {
    int l  = threadIdx.x & 31;
    int m  = l & 15;
    int kb = (l >> 4) << 3;                 // 0 or 8
    const unsigned short* p = base + (size_t)m * ld + k0 + kb;
    Frag32B f;
    f.u[0] = *(const uint4*)(p);
    f.u[1] = *(const uint4*)(p + 16);
    return f.v;
}

// B-fragment (32x16, K x N) from a transposed buffer BT[N][K]:
// lane l -> col n=l&15 (row n of BT); K chunks {kb..kb+7, kb+16..kb+23}, kb=(l>=16)?8:0.
__device__ __forceinline__ v16bf load_fragB(const unsigned short* baseT, int ld, int k0) {
    int l  = threadIdx.x & 31;
    int n  = l & 15;
    int kb = (l >> 4) << 3;                 // 0 or 8
    const unsigned short* p = baseT + (size_t)n * ld + k0 + kb;
    Frag32B f;
    f.u[0] = *(const uint4*)(p);
    f.u[1] = *(const uint4*)(p + 16);
    return f.v;
}

// --------------------------- weight packing --------------------------------
// WQKVT[n][k] = bf16(w_qkv[k][n]) (768x256), WOT[n][k] = bf16(w_o[k][n]) (256x256)
__global__ __launch_bounds__(256) void k_pack_w(const float* __restrict__ wqkv,
                                                const float* __restrict__ wo,
                                                unsigned short* __restrict__ WQKVT,
                                                unsigned short* __restrict__ WOT) {
    int tid = blockIdx.x * 256 + threadIdx.x;
    const int NW1 = 768 * 256;
    if (tid < NW1) {
        int n = tid >> 8, k = tid & 255;
        WQKVT[tid] = f2bf(wqkv[k * 768 + n]);
    } else {
        int t = tid - NW1;
        int n = t >> 8, k = t & 255;
        WOT[t] = f2bf(wo[k * 256 + n]);
    }
}

// ----------------------- window pack of x (bf16) ---------------------------
// XW[g][c], g = (wy*7+wx)*256 + iy*16+ix, (y,x)=(wy*16+iy, wx*16+ix)
__global__ __launch_bounds__(256) void k_pack_x(const float* __restrict__ xin,
                                                unsigned short* __restrict__ XW) {
    int tid = blockIdx.x * 256 + threadIdx.x;
    int g = tid >> 8, c = tid & 255;
    int p = g >> 8, q = g & 255;
    int y = (p / 7) * 16 + (q >> 4);
    int x = (p % 7) * 16 + (q & 15);
    XW[tid] = f2bf(xin[((size_t)(y * IMG + x)) * DIMC + c]);
}

// ------------------------------ QKV GEMM -----------------------------------
// M=12544, N=768, K=256.  32x32 output tile per wave: 4 WMMAs / K-step.
// Epilogue scatters: Q (pre-scaled by QK^-0.5) bf16, K bf16, per-window V^T
// bf16 (VT[p][c][q]) and fp32 V image for LePE.
__global__ __launch_bounds__(128) void k_qkv_gemm(const unsigned short* __restrict__ XW,
                                                  const unsigned short* __restrict__ WQKVT,
                                                  const float* __restrict__ bqkv,
                                                  unsigned short* __restrict__ Qb,
                                                  unsigned short* __restrict__ Kb,
                                                  unsigned short* __restrict__ VT,
                                                  float* __restrict__ Vimg) {
    int wave = blockIdx.x * 4 + (threadIdx.x >> 5);   // 392*24 = 9408 waves
    int mt = wave / 24, nt = wave % 24;               // 32-row x 32-col tiles
    int l = threadIdx.x & 31;
    float bv0 = bqkv[nt * 32 + (l & 15)];
    float bv1 = bqkv[nt * 32 + 16 + (l & 15)];
    v8f acc[2][2];
#pragma unroll
    for (int r = 0; r < 8; ++r) {
        acc[0][0][r] = bv0; acc[1][0][r] = bv0;
        acc[0][1][r] = bv1; acc[1][1][r] = bv1;
    }
    const unsigned short* A0 = XW    + (size_t)(mt * 32) * DIMC;
    const unsigned short* A1 = A0 + 16 * DIMC;
    const unsigned short* B0 = WQKVT + (size_t)(nt * 32) * DIMC;
    const unsigned short* B1 = B0 + 16 * DIMC;
#pragma unroll
    for (int kk = 0; kk < 8; ++kk) {
        if (kk < 7) {
            __builtin_prefetch(A0 + (kk + 1) * 32, 0, 3);
            __builtin_prefetch(A1 + (kk + 1) * 32, 0, 3);
        }
        v16bf a0 = load_fragA(A0, DIMC, kk * 32);
        v16bf a1 = load_fragA(A1, DIMC, kk * 32);
        v16bf b0 = load_fragB(B0, DIMC, kk * 32);
        v16bf b1 = load_fragB(B1, DIMC, kk * 32);
        acc[0][0] = wmma_bf16(a0, b0, acc[0][0]);
        acc[0][1] = wmma_bf16(a0, b1, acc[0][1]);
        acc[1][0] = wmma_bf16(a1, b0, acc[1][0]);
        acc[1][1] = wmma_bf16(a1, b1, acc[1][1]);
    }
    int half8 = (l >> 4) * 8;
#pragma unroll
    for (int i = 0; i < 2; ++i)
#pragma unroll
        for (int j = 0; j < 2; ++j)
#pragma unroll
            for (int r = 0; r < 8; ++r) {
                int row = mt * 32 + i * 16 + r + half8;
                int col = nt * 32 + j * 16 + (l & 15);
                float v = acc[i][j][r];
                if (col < 256) {
                    Qb[(size_t)row * DIMC + col] = f2bf(v * 0.0625f);  // 256^-0.5
                } else if (col < 512) {
                    Kb[(size_t)row * DIMC + (col - 256)] = f2bf(v);
                } else {
                    int c = col - 512;
                    int p = row >> 8, q = row & 255;
                    VT[((size_t)(p * 256 + c)) * 256 + q] = f2bf(v);
                    int y = (p / 7) * 16 + (q >> 4);
                    int x = (p % 7) * 16 + (q & 15);
                    Vimg[((size_t)(y * IMG + x)) * DIMC + c] = v;
                }
            }
}

// ---------------------------- window means ---------------------------------
__global__ __launch_bounds__(256) void k_win_mean(const unsigned short* __restrict__ Qb,
                                                  const unsigned short* __restrict__ Kb,
                                                  float* __restrict__ qwin,
                                                  float* __restrict__ kwin) {
    int b = blockIdx.x;              // 0..97
    int p = b % P2C;
    const unsigned short* src = (b < P2C) ? Qb : Kb;
    float* dst = (b < P2C) ? qwin : kwin;
    int c = threadIdx.x;
    float s = 0.f;
    for (int q = 0; q < 256; ++q)
        s += bf2f(src[((size_t)(p * 256 + q)) * DIMC + c]);
    dst[p * DIMC + c] = s * (1.0f / 256.0f);
}

// ------------------------------ routing ------------------------------------
__global__ __launch_bounds__(64) void k_routing(const float* __restrict__ qwin,
                                                const float* __restrict__ kwin,
                                                int* __restrict__ topidx) {
    __shared__ float lg[64];
    int p = blockIdx.x;
    int j = threadIdx.x;
    if (j < P2C) {
        float s = 0.f;
        for (int c = 0; c < 256; ++c) s += qwin[p * 256 + c] * kwin[j * 256 + c];
        lg[j] = s;
    }
    __syncthreads();
    if (j == 0) {
        for (int t = 0; t < 4; ++t) {
            int best = 0; float bvv = -1e30f;
            for (int w = 0; w < P2C; ++w)
                if (lg[w] > bvv) { bvv = lg[w]; best = w; }
            topidx[p * 4 + t] = best;
            lg[best] = -1e31f;
        }
    }
}

// ------------------------- flash attention (WMMA) --------------------------
// One wave per (window p, head h, 16-query tile qt). 1024 gathered keys
// processed 32 at a time: S = 2 WMMAs, online softmax, P re-fragmented via
// LDS, O += 2 WMMAs against per-window V^T.  Q fragment is reused across all
// 64 S WMMAs of the wave (2 per 32-key step).
__global__ __launch_bounds__(32) void k_attn(const unsigned short* __restrict__ Qb,
                                             const unsigned short* __restrict__ Kb,
                                             const unsigned short* __restrict__ VT,
                                             const int* __restrict__ topidx,
                                             float* __restrict__ OUT) {
    __shared__ __align__(16) unsigned short ldsP[16][32];
    int b  = blockIdx.x;
    int p  = b >> 7;
    int h  = (b >> 4) & 7;
    int qt = b & 15;
    int l  = threadIdx.x & 31;
    int half8 = (l >> 4) * 8;

    int sel[4];
#pragma unroll
    for (int i = 0; i < 4; ++i) sel[i] = topidx[p * 4 + i];

    const unsigned short* Qrow = Qb + ((size_t)(p * 256 + qt * 16)) * DIMC + h * 32;
    v16bf qf = load_fragA(Qrow, DIMC, 0);

    float rm[8], rs[8];
    v8f o0 = {}, o1 = {};
#pragma unroll
    for (int r = 0; r < 8; ++r) { rm[r] = -1e30f; rs[r] = 0.f; }

    for (int kt2 = 0; kt2 < 32; ++kt2) {
        int sw  = sel[kt2 >> 3];
        int qk0 = (kt2 & 7) * 32;

        const unsigned short* Kbase = Kb + ((size_t)(sw * 256 + qk0)) * DIMC + h * 32;
        v16bf kb0 = load_fragB(Kbase,             DIMC, 0);   // keys qk0+n
        v16bf kb1 = load_fragB(Kbase + 16 * DIMC, DIMC, 0);   // keys qk0+16+n
        v8f z = {};
        v8f s0 = wmma_bf16(qf, kb0, z);
        v8f s1 = wmma_bf16(qf, kb1, z);

#pragma unroll
        for (int r = 0; r < 8; ++r) {
            float mx = fmaxf(s0[r], s1[r]);
#pragma unroll
            for (int off = 1; off < 16; off <<= 1)
                mx = fmaxf(mx, __shfl_xor(mx, off, 16));
            float mnew = fmaxf(rm[r], mx);
            float e0 = __expf(s0[r] - mnew);
            float e1 = __expf(s1[r] - mnew);
            float ts = e0 + e1;
#pragma unroll
            for (int off = 1; off < 16; off <<= 1)
                ts += __shfl_xor(ts, off, 16);
            float alpha = __expf(rm[r] - mnew);
            rs[r] = rs[r] * alpha + ts;
            rm[r] = mnew;
            o0[r] *= alpha;
            o1[r] *= alpha;
            ldsP[r + half8][l & 15]        = f2bf(e0);
            ldsP[r + half8][16 + (l & 15)] = f2bf(e1);
        }
        __syncthreads();

        v16bf pf = load_fragA(&ldsP[0][0], 32, 0);
        const unsigned short* Vbase = VT + ((size_t)(sw * 256 + h * 32)) * 256 + qk0;
        v16bf v0 = load_fragB(Vbase,            256, 0);   // out cols d = n
        v16bf v1 = load_fragB(Vbase + 16 * 256, 256, 0);   // out cols d = n+16
        o0 = wmma_bf16(pf, v0, o0);
        o1 = wmma_bf16(pf, v1, o1);
        __syncthreads();
    }

#pragma unroll
    for (int r = 0; r < 8; ++r) {
        float inv = 1.0f / rs[r];
        int row = p * 256 + qt * 16 + r + half8;
        size_t base = (size_t)row * DIMC + h * 32;
        OUT[base + (l & 15)]      = o0[r] * inv;
        OUT[base + 16 + (l & 15)] = o1[r] * inv;
    }
}

// ------------------- LePE (depthwise 3x3) + fuse + pack --------------------
__global__ __launch_bounds__(256) void k_lepe_fuse(const float* __restrict__ Vimg,
                                                   const float* __restrict__ attn,
                                                   const float* __restrict__ lw,
                                                   const float* __restrict__ lb,
                                                   unsigned short* __restrict__ AFIN) {
    int pix = blockIdx.x;            // image order
    int c = threadIdx.x;
    int y = pix / IMG, x = pix % IMG;
    float acc = lb[c];
#pragma unroll
    for (int dy = 0; dy < 3; ++dy)
#pragma unroll
        for (int dx = 0; dx < 3; ++dx) {
            int yy = y + dy - 1, xx = x + dx - 1;
            if (yy >= 0 && yy < IMG && xx >= 0 && xx < IMG)
                acc += Vimg[((size_t)(yy * IMG + xx)) * DIMC + c] * lw[(dy * 3 + dx) * DIMC + c];
        }
    int wy = y >> 4, iy = y & 15, wx = x >> 4, ix = x & 15;
    int g = (wy * 7 + wx) * 256 + iy * 16 + ix;
    acc += attn[(size_t)g * DIMC + c];
    AFIN[(size_t)pix * DIMC + c] = f2bf(acc);
}

// ---------------------------- output GEMM ----------------------------------
// d_out = AFIN(12544x256) @ w_o(256x256) + b_o, fp32 out in image order.
// 32x32 output tile per wave.
__global__ __launch_bounds__(128) void k_out_gemm(const unsigned short* __restrict__ AFIN,
                                                  const unsigned short* __restrict__ WOT,
                                                  const float* __restrict__ bo,
                                                  float* __restrict__ out) {
    int wave = blockIdx.x * 4 + (threadIdx.x >> 5);   // 392*8 = 3136 waves
    int mt = wave >> 3, nt = wave & 7;                // 32-row x 32-col tiles
    int l = threadIdx.x & 31;
    float bv0 = bo[nt * 32 + (l & 15)];
    float bv1 = bo[nt * 32 + 16 + (l & 15)];
    v8f acc[2][2];
#pragma unroll
    for (int r = 0; r < 8; ++r) {
        acc[0][0][r] = bv0; acc[1][0][r] = bv0;
        acc[0][1][r] = bv1; acc[1][1][r] = bv1;
    }
    const unsigned short* A0 = AFIN + (size_t)(mt * 32) * DIMC;
    const unsigned short* A1 = A0 + 16 * DIMC;
    const unsigned short* B0 = WOT  + (size_t)(nt * 32) * DIMC;
    const unsigned short* B1 = B0 + 16 * DIMC;
#pragma unroll
    for (int kk = 0; kk < 8; ++kk) {
        if (kk < 7) {
            __builtin_prefetch(A0 + (kk + 1) * 32, 0, 3);
            __builtin_prefetch(A1 + (kk + 1) * 32, 0, 3);
        }
        v16bf a0 = load_fragA(A0, DIMC, kk * 32);
        v16bf a1 = load_fragA(A1, DIMC, kk * 32);
        v16bf b0 = load_fragB(B0, DIMC, kk * 32);
        v16bf b1 = load_fragB(B1, DIMC, kk * 32);
        acc[0][0] = wmma_bf16(a0, b0, acc[0][0]);
        acc[0][1] = wmma_bf16(a0, b1, acc[0][1]);
        acc[1][0] = wmma_bf16(a1, b0, acc[1][0]);
        acc[1][1] = wmma_bf16(a1, b1, acc[1][1]);
    }
    int half8 = (l >> 4) * 8;
#pragma unroll
    for (int i = 0; i < 2; ++i)
#pragma unroll
        for (int j = 0; j < 2; ++j)
#pragma unroll
            for (int r = 0; r < 8; ++r)
                out[(size_t)(mt * 32 + i * 16 + r + half8) * DIMC
                    + nt * 32 + j * 16 + (l & 15)] = acc[i][j][r];
}

// ---------------------------------------------------------------------------
extern "C" void kernel_launch(void* const* d_in, const int* in_sizes, int n_in,
                              void* d_out, int out_size, void* d_ws, size_t ws_size,
                              hipStream_t stream) {
    const float* x    = (const float*)d_in[0];
    const float* wqkv = (const float*)d_in[1];
    const float* bqkv = (const float*)d_in[2];
    const float* wo   = (const float*)d_in[3];
    const float* bo   = (const float*)d_in[4];
    const float* lw   = (const float*)d_in[5];
    const float* lb   = (const float*)d_in[6];
    float* out = (float*)d_out;

    char* ws = (char*)d_ws;
    size_t off = 0;
    auto carve = [&](size_t bytes) -> char* {
        char* p = ws + off;
        off = (off + bytes + 255) & ~(size_t)255;
        return p;
    };
    unsigned short* XW    = (unsigned short*)carve((size_t)NPIX * DIMC * 2);
    unsigned short* WQKVT = (unsigned short*)carve((size_t)768 * 256 * 2);
    unsigned short* WOT   = (unsigned short*)carve((size_t)256 * 256 * 2);
    unsigned short* Qb    = (unsigned short*)carve((size_t)NPIX * DIMC * 2);
    unsigned short* Kb    = (unsigned short*)carve((size_t)NPIX * DIMC * 2);
    unsigned short* VT    = (unsigned short*)carve((size_t)NPIX * DIMC * 2);
    float*          Vimg  = (float*)carve((size_t)NPIX * DIMC * 4);
    float*          qwin  = (float*)carve((size_t)P2C * DIMC * 4);
    float*          kwin  = (float*)carve((size_t)P2C * DIMC * 4);
    int*            tidx  = (int*)carve((size_t)P2C * 4 * 4);
    float*          ATTN  = (float*)carve((size_t)NPIX * DIMC * 4);
    unsigned short* AFIN  = (unsigned short*)carve((size_t)NPIX * DIMC * 2);

    k_pack_w   <<<(768*256 + 256*256) / 256, 256, 0, stream>>>(wqkv, wo, WQKVT, WOT);
    k_pack_x   <<<NPIX, 256, 0, stream>>>(x, XW);
    k_qkv_gemm <<<(392 * 24) / 4, 128, 0, stream>>>(XW, WQKVT, bqkv, Qb, Kb, VT, Vimg);
    k_win_mean <<<2 * P2C, 256, 0, stream>>>(Qb, Kb, qwin, kwin);
    k_routing  <<<P2C, 64, 0, stream>>>(qwin, kwin, tidx);
    k_attn     <<<P2C * 8 * 16, 32, 0, stream>>>(Qb, Kb, VT, tidx, ATTN);
    k_lepe_fuse<<<NPIX, 256, 0, stream>>>(Vimg, ATTN, lw, lb, AFIN);
    k_out_gemm <<<(392 * 8) / 4, 128, 0, stream>>>(AFIN, WOT, bo, out);
}
